// HashGridNeRF_64149631533764
// MI455X (gfx1250) — matrix operand order repaired and checked by reading, not compiled, not run on hardware
//
#include <hip/hip_runtime.h>

// ---------------------------------------------------------------------------
// Instant-NGP hash-grid encode + fused 32->64->64->1 MLP for gfx1250 (MI455X)
//
//  * Each wave (32 lanes) owns a 16-point tile; block = 4 waves = 64 points.
//  * Encode: lane = (point 0..15) x (level-half 0/1); 8 levels/lane, 8
//    float2 gathers per level. Tables are 64 MB total -> resident in the
//    192 MB L2, so gathers are L2-bandwidth, not HBM, bound.
//  * Per-level axis terms (hash products / clamped strides / weight outer
//    products) are hoisted so each corner costs ~6 VALU ops + one 64-bit load.
//  * MLP: V_WMMA_F32_16X16X32_F16: 4 WMMAs (enc@W_in) + 8 WMMAs (h1@W_hid)
//    per wave tile; 64->1 output layer as a lane-split dot + shfl reduce.
// ---------------------------------------------------------------------------

typedef __attribute__((ext_vector_type(16))) _Float16 v16h;
typedef __attribute__((ext_vector_type(8)))  _Float16 v8h;
typedef __attribute__((ext_vector_type(8)))  float    v8f;

#define NLV       16
#define TBLSZ     (1u << 19)      // 524288 entries/level (power of two -> mask)
#define HID       64
#define ENCD      32
#define WAVES     4
#define PTS_BLK   64              // 4 waves * 16 points

#define PRIME1    2654435761u
#define PRIME2    805459861u

// scale_l = 16 * 1.4472692012786865^l - 1 ; res_l = ceil(scale_l)+1
__constant__ float c_scale[NLV] = {
    15.0f,        22.1563072f,  32.5134125f,  47.5029373f,
    69.1968155f,  100.593720f,  146.033508f,  211.796936f,
    306.974640f,  444.722320f,  644.080200f,  932.604700f,
    1350.17735f,  1954.51735f,  2829.16000f,  4095.00340f};
__constant__ int c_res[NLV] = {
    16, 24, 34, 49, 71, 102, 148, 213, 308, 446, 646, 934, 1352, 1956, 2831, 4097};

__device__ __forceinline__ v8h ld8h(const _Float16* p) {
    return *(const v8h*)p;
}
__device__ __forceinline__ float relu(float x) { return x > 0.0f ? x : 0.0f; }
__device__ __forceinline__ int clampi(int v, int hi) {
    return v < 0 ? 0 : (v > hi ? hi : v);
}

union V16 { v16h v; v8h h[2]; };

__global__ __launch_bounds__(128)
void hashgrid_nerf_fused(const float* __restrict__ coords,
                         const float* __restrict__ tables,
                         const float* __restrict__ Win,
                         const float* __restrict__ Whid,
                         const float* __restrict__ Wout,
                         float* __restrict__ out, int P)
{
    // weights, f16, transposed to [n][k] so B fragments are contiguous 32B
    __shared__ __align__(32) _Float16 sWinT [HID * ENCD];        // 4 KB
    __shared__ __align__(32) _Float16 sWhidT[HID * HID];         // 8 KB
    __shared__ float                  sWout [HID];               // 256 B
    // per-wave staging (padded against LDS bank conflicts)
    __shared__ __align__(16) _Float16 sEnc[WAVES][16][ENCD + 8]; // 5 KB
    __shared__ __align__(16) _Float16 sH1 [WAVES][16][HID  + 8]; // 9 KB
    __shared__ float                  sH2 [WAVES][16][HID  + 1]; // 16.25 KB

    const int tid  = threadIdx.x;
    const int lane = tid & 31;
    const int wv   = tid >> 5;     // wave in block: 0..3
    const int row  = lane & 15;    // point within wave tile
    const int half = lane >> 4;    // 0: levels 0-7 / K-low ; 1: levels 8-15 / K-high

    // ---- stage weights to LDS as f16, transposed ----
    for (int i = tid; i < ENCD * HID; i += blockDim.x) {
        int k = i >> 6, n = i & 63;                  // Win is (32,64) row-major
        sWinT[n * ENCD + k] = (_Float16)Win[i];
    }
    for (int i = tid; i < HID * HID; i += blockDim.x) {
        int k = i >> 6, n = i & 63;                  // Whid is (64,64) row-major
        sWhidT[n * HID + k] = (_Float16)Whid[i];
    }
    if (tid < HID) sWout[tid] = Wout[tid];

    // ---- hash-grid encode: 8 levels per lane, one point per lane-pair ----
    long pt = (long)blockIdx.x * PTS_BLK + wv * 16 + row;
    const bool valid = pt < (long)P;
    const long cpt = valid ? pt : (long)(P - 1);
    const float x = coords[cpt * 3 + 0];
    const float y = coords[cpt * 3 + 1];
    const float z = coords[cpt * 3 + 2];

    #pragma unroll
    for (int li = 0; li < 8; ++li) {
        const int   lvl   = half * 8 + li;
        const float scale = c_scale[lvl];
        const float* tbl  = tables + (size_t)lvl * (TBLSZ * 2u);

        const float px = x * scale + 0.5f;
        const float py = y * scale + 0.5f;
        const float pz = z * scale + 0.5f;
        const float bx = floorf(px), by = floorf(py), bz = floorf(pz);
        const float fx = px - bx, fy = py - by, fz = pz - bz;
        const int ix = (int)bx, iy = (int)by, iz = (int)bz;

        // trilinear weight outer products (hoisted: 1 mul per corner in loop)
        const float wx0 = 1.0f - fx, wy0 = 1.0f - fy;
        const float wz_[2]  = {1.0f - fz, fz};
        const float wxy[4]  = {wx0 * wy0, fx * wy0, wx0 * fy, fx * fy};

        float a0 = 0.0f, a1 = 0.0f;
        if (lvl < 5) {
            // ---- dense level: clamped strided index, additive per corner ----
            const int res  = c_res[lvl];
            const int res2 = res * res;
            const unsigned ax[2] = {(unsigned)clampi(ix, res - 1),
                                    (unsigned)clampi(ix + 1, res - 1)};
            const unsigned ay[2] = {(unsigned)(clampi(iy, res - 1) * res),
                                    (unsigned)(clampi(iy + 1, res - 1) * res)};
            const unsigned az[2] = {(unsigned)(clampi(iz, res - 1) * res2),
                                    (unsigned)(clampi(iz + 1, res - 1) * res2)};
            #pragma unroll
            for (int c = 0; c < 8; ++c) {
                const unsigned idx = ax[c & 1] + ay[(c >> 1) & 1] + az[c >> 2];
                const float2 t = *(const float2*)(tbl + 2u * idx);
                const float w = wxy[c & 3] * wz_[c >> 2];
                a0 = fmaf(w, t.x, a0);
                a1 = fmaf(w, t.y, a1);
            }
        } else {
            // ---- hashed level: separable xor hash, 2 xor + and per corner ----
            const unsigned ax[2] = {(unsigned)ix, (unsigned)(ix + 1)};
            const unsigned ay[2] = {(unsigned)iy * PRIME1,
                                    (unsigned)(iy + 1) * PRIME1};
            const unsigned az[2] = {(unsigned)iz * PRIME2,
                                    (unsigned)(iz + 1) * PRIME2};
            #pragma unroll
            for (int c = 0; c < 8; ++c) {
                const unsigned idx =
                    (ax[c & 1] ^ ay[(c >> 1) & 1] ^ az[c >> 2]) & (TBLSZ - 1u);
                const float2 t = *(const float2*)(tbl + 2u * idx);
                const float w = wxy[c & 3] * wz_[c >> 2];
                a0 = fmaf(w, t.x, a0);
                a1 = fmaf(w, t.y, a1);
            }
        }
        sEnc[wv][row][2 * lvl + 0] = (_Float16)a0;
        sEnc[wv][row][2 * lvl + 1] = (_Float16)a1;
    }
    __syncthreads();

    // ---- layer 1: enc(16x32) @ W_in(32x64) -> 4 WMMA tiles ----
    // A k order chosen here (contiguous, matches B): lanes 0-15 hold K 0-15, lanes 16-31 K 16-31
    V16 aEnc;
    aEnc.h[0] = ld8h(&sEnc[wv][row][half * 16]);
    aEnc.h[1] = ld8h(&sEnc[wv][row][half * 16 + 8]);
    const int col = row;           // B/D column within 16-wide tile
    const int kb  = half * 16;     // B layout: lanes 0-15 K 0-15, lanes 16-31 K 16-31

    v8f acc[4];
    #pragma unroll
    for (int n = 0; n < 4; ++n) {
        const v16h b = *(const v16h*)&sWinT[(n * 16 + col) * ENCD + kb];
        v8f cz = {};
        acc[n] = __builtin_amdgcn_wmma_f32_16x16x32_f16(
            false, aEnc.v, false, b, (short)0, cz, false, false);
    }
    // ReLU + f16 -> LDS (C/D layout: VGPR r <-> M = r + 8*half, N = col)
    #pragma unroll
    for (int n = 0; n < 4; ++n)
        #pragma unroll
        for (int r = 0; r < 8; ++r)
            sH1[wv][r + half * 8][n * 16 + col] = (_Float16)relu(acc[n][r]);
    __syncthreads();

    // ---- layer 2: h1(16x64) @ W_hid(64x64) -> 8 WMMAs (K split in two) ----
    V16 aLo, aHi;
    aLo.h[0] = ld8h(&sH1[wv][row][half * 16]);
    aLo.h[1] = ld8h(&sH1[wv][row][half * 16 + 8]);
    aHi.h[0] = ld8h(&sH1[wv][row][32 + half * 16]);
    aHi.h[1] = ld8h(&sH1[wv][row][32 + half * 16 + 8]);
    #pragma unroll
    for (int n = 0; n < 4; ++n) {
        const v16h b0 = *(const v16h*)&sWhidT[(n * 16 + col) * HID + kb];
        const v16h b1 = *(const v16h*)&sWhidT[(n * 16 + col) * HID + 32 + kb];
        v8f c = {};
        c = __builtin_amdgcn_wmma_f32_16x16x32_f16(
            false, aLo.v, false, b0, (short)0, c, false, false);
        c = __builtin_amdgcn_wmma_f32_16x16x32_f16(
            false, aHi.v, false, b1, (short)0, c, false, false);
        #pragma unroll
        for (int r = 0; r < 8; ++r)
            sH2[wv][r + half * 8][n * 16 + col] = relu(c[r]);
    }
    __syncthreads();

    // ---- output layer: dot(h2[row], W_out), split across lane-halves ----
    float part = 0.0f;
    const float* hr = &sH2[wv][row][half * 32];
    const float* wo = &sWout[half * 32];
    #pragma unroll
    for (int j = 0; j < 32; ++j) part = fmaf(hr[j], wo[j], part);
    part += __shfl_xor(part, 16, 32);
    if (valid && half == 0) out[pt] = part;
}

extern "C" void kernel_launch(void* const* d_in, const int* in_sizes, int n_in,
                              void* d_out, int out_size, void* d_ws, size_t ws_size,
                              hipStream_t stream) {
    (void)n_in; (void)out_size; (void)d_ws; (void)ws_size;
    const float* coords = (const float*)d_in[0];   // (4,512,512,3) f32
    const float* tables = (const float*)d_in[1];   // (16, 524288, 2) f32
    const float* Win    = (const float*)d_in[2];   // (32,64) f32
    const float* Whid   = (const float*)d_in[3];   // (64,64) f32
    const float* Wout   = (const float*)d_in[4];   // (64,1)  f32
    float* out = (float*)d_out;                    // (P,) f32

    const int P = in_sizes[0] / 3;
    const int grid = (P + PTS_BLK - 1) / PTS_BLK;
    hipLaunchKernelGGL(hashgrid_nerf_fused, dim3(grid), dim3(128), 0, stream,
                       coords, tables, Win, Whid, Wout, out, P);
}
